// SpikeSparseConnectomeRoutingAttention_61306363183703
// MI455X (gfx1250) — compile-verified, matrix operand order checked by reading, not run on hardware
//
#include <hip/hip_runtime.h>
#include <hip/hip_bf16.h>

// ---- problem constants (from reference) ----
#define BB 2
#define NN 2048
#define DD 1024
#define HH 16
#define CC 16
#define KK 64
#define FF 32
#define DH 64            // DD/HH
#define MM (BB*NN)       // 4096 rows
#define TK 32            // GEMM K-tile

typedef __attribute__((ext_vector_type(16))) __bf16 v16bf;
typedef __attribute__((ext_vector_type(8)))  float  v8f;

__device__ __forceinline__ v8f wmma_bf16(v16bf a, v16bf b, v8f c) {
  return __builtin_amdgcn_wmma_f32_16x16x32_bf16(false, a, false, b, (short)0, c, false, false);
}

// Async global->LDS copy of 16 bytes per lane (CDNA5, tracked by ASYNCcnt).
// vdst = wave-relative LDS byte address, vaddr = 64-bit global address.
__device__ __forceinline__ void async_copy16(unsigned lds_byte_off, const __bf16* g) {
  asm volatile("global_load_async_to_lds_b128 %0, %1, off"
               :: "v"(lds_byte_off), "v"(g)
               : "memory");
}

// ---------------------------------------------------------------------------
// f32 -> bf16 conversion
// ---------------------------------------------------------------------------
__global__ __launch_bounds__(256) void f32_to_bf16_kernel(const float* __restrict__ src,
                                                          __bf16* __restrict__ dst, int n) {
  int i = blockIdx.x * 256 + threadIdx.x;
  if (i < n) dst[i] = (__bf16)src[i];
}

// ---------------------------------------------------------------------------
// zero fill
// ---------------------------------------------------------------------------
__global__ __launch_bounds__(256) void zero_f32_kernel(float* __restrict__ p, int n) {
  int i = blockIdx.x * 256 + threadIdx.x;
  if (i < n) p[i] = 0.0f;
}

// ---------------------------------------------------------------------------
// prep: RMSNorm row -> bf16, rope_add row (f32)
// one block per row (s*N+n), 256 threads
// ---------------------------------------------------------------------------
__global__ __launch_bounds__(256)
void prep_kernel(const float* __restrict__ x, const float* __restrict__ positions,
                 const float* __restrict__ w_rms, const float* __restrict__ rope_dirs,
                 const float* __restrict__ rope_freqs, const float* __restrict__ w_rope,
                 __bf16* __restrict__ xn, float* __restrict__ rope_add) {
  const int row = blockIdx.x;
  const int tid = threadIdx.x;
  __shared__ float red[256];
  __shared__ float emb[2 * FF];
  __shared__ float rms_s;

  const float* xr = x + (size_t)row * DD;
  float ss = 0.f;
  for (int i = tid; i < DD; i += 256) { float xv = xr[i]; ss += xv * xv; }
  red[tid] = ss;
  __syncthreads();
  for (int st = 128; st > 0; st >>= 1) { if (tid < st) red[tid] += red[tid + st]; __syncthreads(); }
  if (tid == 0) rms_s = rsqrtf(red[0] / (float)DD + 1e-6f);

  if (tid < FF) {
    const float* pp = positions + (size_t)row * 3;
    float ang = (pp[0] * rope_dirs[tid * 3 + 0] + pp[1] * rope_dirs[tid * 3 + 1] +
                 pp[2] * rope_dirs[tid * 3 + 2]) * rope_freqs[tid];
    emb[tid] = __sinf(ang);
    emb[FF + tid] = __cosf(ang);
  }
  __syncthreads();

  const float rr = rms_s;
  for (int i = tid; i < DD; i += 256)
    xn[(size_t)row * DD + i] = (__bf16)(xr[i] * rr * w_rms[i]);

  for (int col = tid; col < DD; col += 256) {
    const float* wr = w_rope + (size_t)col * (2 * FF);
    float acc = 0.f;
    #pragma unroll
    for (int j = 0; j < 2 * FF; ++j) acc += emb[j] * wr[j];
    rope_add[(size_t)row * DD + col] = acc;
  }
}

// ---------------------------------------------------------------------------
// bf16 WMMA GEMM:  C(M,N) = A(M,K) * Bt(N,K)^T  [+ bias(M,N)]
// block: 256 threads (8 waves), 128x128 tile; wave: 32x64 (2x4 subtiles)
// A/B K-tiles staged in LDS via async global->LDS DMA, double buffered.
// M,N multiples of 128; K multiple of 32
// ---------------------------------------------------------------------------
__global__ __launch_bounds__(256)
void gemm_bf16_kernel(const __bf16* __restrict__ A, const __bf16* __restrict__ Bt,
                      const float* __restrict__ bias, float* __restrict__ Cmat,
                      int Mdim, int Ndim, int Kdim) {
  __shared__ __attribute__((aligned(32))) __bf16 Abuf[2][128 * TK];
  __shared__ __attribute__((aligned(32))) __bf16 Bbuf[2][128 * TK];

  const int tid  = threadIdx.x;
  const int lane = tid & 31;
  const int wave = tid >> 5;
  const int wr = wave >> 1;                // 0..3
  const int wc = wave & 1;                 // 0..1
  const int bm = blockIdx.x * 128;
  const int bn = blockIdx.y * 128;
  const int lrow = lane & 15;
  const int kh = (lane >> 4) * 16;

  v8f acc[2][4] = {};
  const int steps = Kdim / TK;

  // Stage one 128xTK A tile and one 128xTK B tile into LDS buffer `buf`.
  // 512 chunks of 16B per tile; thread handles chunk indices tid and tid+256.
  // Per wave: 4 async instructions outstanding per issued stage.
  auto issue = [&](int buf, int k0) {
    #pragma unroll
    for (int e0 = 0; e0 < 2; ++e0) {
      int e = tid + e0 * 256;              // 0..511
      int row = e >> 2;                    // 0..127
      int chunk = e & 3;                   // 0..3 (8 bf16 each)
      unsigned lofsA = (unsigned)(size_t)&Abuf[buf][row * TK + chunk * 8];
      unsigned lofsB = (unsigned)(size_t)&Bbuf[buf][row * TK + chunk * 8];
      async_copy16(lofsA, A + (size_t)(bm + row) * Kdim + k0 + chunk * 8);
      async_copy16(lofsB, Bt + (size_t)(bn + row) * Kdim + k0 + chunk * 8);
    }
  };

  issue(0, 0);
  for (int i = 0; i < steps; ++i) {
    const int cur = i & 1;
    const bool have_next = (i + 1) < steps;
    if (have_next) {
      issue(cur ^ 1, (i + 1) * TK);
      asm volatile("s_wait_asynccnt 0x4" ::: "memory");  // step-i DMAs done
    } else {
      asm volatile("s_wait_asynccnt 0x0" ::: "memory");
    }
    __syncthreads();

    v16bf a0 = *(const v16bf*)&Abuf[cur][(wr * 32 + lrow) * TK + kh];
    v16bf a1 = *(const v16bf*)&Abuf[cur][(wr * 32 + 16 + lrow) * TK + kh];
    #pragma unroll
    for (int nt = 0; nt < 4; ++nt) {
      v16bf b = *(const v16bf*)&Bbuf[cur][(wc * 64 + nt * 16 + lrow) * TK + kh];
      acc[0][nt] = wmma_bf16(a0, b, acc[0][nt]);
      acc[1][nt] = wmma_bf16(a1, b, acc[1][nt]);
    }
    __syncthreads();
  }

  const int crow_base = bm + wr * 32 + ((lane >> 4) ? 8 : 0);
  const int ccol_base = bn + wc * 64 + lrow;
  #pragma unroll
  for (int mt = 0; mt < 2; ++mt)
    #pragma unroll
    for (int nt = 0; nt < 4; ++nt)
      #pragma unroll
      for (int r = 0; r < 8; ++r) {
        int rowi = crow_base + mt * 16 + r;
        int coli = ccol_base + nt * 16;
        size_t idx = (size_t)rowi * Ndim + coli;
        float v = acc[mt][nt][r];
        if (bias) v += bias[idx];
        Cmat[idx] = v;
      }
}

// ---------------------------------------------------------------------------
// cluster scores + top-K selection
// one block per (s,h,c); 256 threads; t = q or k (M,D), cent = (H*C, 67)
// ---------------------------------------------------------------------------
__global__ __launch_bounds__(256)
void score_topk_kernel(const float* __restrict__ t, const float* __restrict__ positions,
                       const float* __restrict__ cent, int* __restrict__ idx_out) {
  const int bid = blockIdx.x;
  const int c = bid % CC;
  const int h = (bid / CC) % HH;
  const int s = bid / (CC * HH);
  const int tid = threadIdx.x;

  __shared__ unsigned long long keys[NN];
  __shared__ unsigned long long red[256];
  __shared__ float cvec[68];
  __shared__ float cnorm_s;

  const float* crow = cent + (size_t)(h * CC + c) * (DH + 3);
  if (tid < DH + 3) cvec[tid] = crow[tid];
  __syncthreads();
  if (tid == 0) {
    float ss = 0.f;
    for (int i = 0; i < DH + 3; ++i) ss += cvec[i] * cvec[i];
    cnorm_s = rsqrtf(fmaxf(ss, 1e-24f));
  }
  __syncthreads();
  const float cn = cnorm_s;

  for (int n = tid; n < NN; n += 256) {
    const float* tr = t + ((size_t)(s * NN + n)) * DD + h * DH;
    float dotf = 0.f, tn2 = 0.f;
    #pragma unroll 8
    for (int d = 0; d < DH; ++d) { float qv = tr[d]; dotf += qv * cvec[d]; tn2 += qv * qv; }
    const float* pp = positions + (size_t)(s * NN + n) * 3;
    float px = pp[0], py = pp[1], pz = pp[2];
    float dotp = px * cvec[DH] + py * cvec[DH + 1] + pz * cvec[DH + 2];
    float pn2 = px * px + py * py + pz * pz;
    float denom = sqrtf(fmaxf(tn2 + pn2, 1e-12f));
    float score = (dotf + dotp) * cn / denom;
    if (px == 0.f && py == 0.f && pz == 0.f) score = 0.f;
    unsigned u = __float_as_uint(score);
    u = (u & 0x80000000u) ? ~u : (u | 0x80000000u);
    keys[n] = ((unsigned long long)u << 32) | (unsigned)n;
  }
  __syncthreads();

  int* out = idx_out + (size_t)bid * KK;
  for (int kk = 0; kk < KK; ++kk) {
    unsigned long long best = 0ull;
    for (int n = tid; n < NN; n += 256) {
      unsigned long long kv = keys[n];
      best = (kv > best) ? kv : best;
    }
    red[tid] = best;
    __syncthreads();
    for (int st = 128; st > 0; st >>= 1) {
      if (tid < st) { unsigned long long o = red[tid + st]; if (o > red[tid]) red[tid] = o; }
      __syncthreads();
    }
    if (tid == 0) {
      unsigned nwin = (unsigned)(red[0] & 0xffffffffu);
      out[kk] = (int)nwin;
      keys[nwin] = 0ull;
    }
    __syncthreads();
  }
}

// ---------------------------------------------------------------------------
// per-cluster attention: gather, WMMA att(64x65), softmax, WMMA o(64x64),
// atomic scatter into acc/cnt. one block per (s,h,c); 128 threads (4 waves).
// ---------------------------------------------------------------------------
__global__ __launch_bounds__(128)
void attention_kernel(const float* __restrict__ q, const float* __restrict__ k,
                      const float* __restrict__ v, const int* __restrict__ in_idx,
                      const int* __restrict__ out_idx, const unsigned char* __restrict__ spike,
                      const float* __restrict__ null_vec,
                      float* __restrict__ acc_out, float* __restrict__ cnt_out) {
  const int bid = blockIdx.x;
  const int h = (bid / CC) % HH;
  const int s = bid / (CC * HH);
  const int tid = threadIdx.x;

  __shared__ int iidx[KK];
  __shared__ int oidx[KK];
  __shared__ float kvkeep[KK + 1];
  __shared__ float s_keep;
  __shared__ __attribute__((aligned(32))) __bf16 qg[64 * 64];
  __shared__ __attribute__((aligned(32))) __bf16 kf[80 * 64];
  __shared__ float att[64 * 80];
  __shared__ __attribute__((aligned(32))) __bf16 pm[64 * 96];
  __shared__ __attribute__((aligned(32))) __bf16 vT[64 * 96];

  const int* ii = in_idx + (size_t)bid * KK;
  const int* oi = out_idx + (size_t)bid * KK;
  if (tid < KK) { iidx[tid] = ii[tid]; oidx[tid] = oi[tid]; }
  __syncthreads();

  if (tid == 0) {
    int any = 0;
    for (int m = 0; m < KK; ++m) {
      int sp = spike[s * NN + oidx[m]] ? 1 : 0;
      kvkeep[m] = (float)sp;
      any |= sp;
    }
    for (int m = 0; m < KK; ++m) kvkeep[m] *= (float)any;
    kvkeep[KK] = (float)any;
    s_keep = (float)any;
  }

  // gather q/k/v head slices into LDS (bf16); vT transposed for 2nd GEMM
  {
    int row = tid >> 1, half = tid & 1;
    const float* qsrc = q + ((size_t)(s * NN + iidx[row])) * DD + h * DH;
    const float* ksrc = k + ((size_t)(s * NN + oidx[row])) * DD + h * DH;
    const float* vsrc = v + ((size_t)(s * NN + oidx[row])) * DD + h * DH;
    for (int j = 0; j < 32; ++j) {
      int d = half * 32 + j;
      qg[row * 64 + d] = (__bf16)qsrc[d];
      kf[row * 64 + d] = (__bf16)ksrc[d];
      vT[d * 96 + row] = (__bf16)vsrc[d];
    }
    if (tid < DH) {
      int d = tid;
      float nv = null_vec[h * DH + d];
      kf[KK * 64 + d] = (__bf16)nv;            // null key row (m = 64)
      for (int rp = KK + 1; rp < 80; ++rp) kf[rp * 64 + d] = (__bf16)0.f;
      vT[d * 96 + KK] = (__bf16)nv;            // null value col in vT
      for (int cp = KK + 1; cp < 96; ++cp) vT[d * 96 + cp] = (__bf16)0.f;
    }
  }
  __syncthreads();

  const int wave = tid >> 5;
  const int lane = tid & 31;
  const int lrow = lane & 15;
  const int kh = (lane >> 4) * 16;

  // GEMM1: att[64 x 80] = qg[64 x 64] * kf^T  (5 N-tiles per wave)
  {
    v8f a1[5] = {};
    for (int k0 = 0; k0 < 64; k0 += 32) {
      v16bf af = *(const v16bf*)(&qg[(wave * 16 + lrow) * 64 + k0 + kh]);
      #pragma unroll
      for (int nt = 0; nt < 5; ++nt) {
        v16bf bf = *(const v16bf*)(&kf[(nt * 16 + lrow) * 64 + k0 + kh]);
        a1[nt] = wmma_bf16(af, bf, a1[nt]);
      }
    }
    #pragma unroll
    for (int nt = 0; nt < 5; ++nt)
      #pragma unroll
      for (int r = 0; r < 8; ++r) {
        int row = wave * 16 + r + ((lane >> 4) ? 8 : 0);
        int col = nt * 16 + lrow;
        att[row * 80 + col] = a1[nt][r];
      }
  }
  __syncthreads();

  // softmax over 65 entries per row (threads 0..63)
  if (tid < 64) {
    float mx = -3.4e38f;
    for (int m = 0; m <= KK; ++m) {
      float vv = (kvkeep[m] != 0.f) ? att[tid * 80 + m] * 0.125f : -1e9f;
      mx = fmaxf(mx, vv);
    }
    float ssum = 0.f;
    for (int m = 0; m <= KK; ++m) {
      float vv = (kvkeep[m] != 0.f) ? att[tid * 80 + m] * 0.125f : -1e9f;
      float e = __expf(vv - mx);
      att[tid * 80 + m] = e;
      ssum += e;
    }
    float inv = 1.f / ssum;
    for (int m = 0; m <= KK; ++m) pm[tid * 96 + m] = (__bf16)(att[tid * 80 + m] * inv);
    for (int m = KK + 1; m < 96; ++m) pm[tid * 96 + m] = (__bf16)0.f;
  }
  __syncthreads();

  // GEMM2: o[64 x 64] = pm[64 x 96] * vT^T  (K padded to 96)
  v8f a2[4] = {};
  for (int k0 = 0; k0 < 96; k0 += 32) {
    v16bf af = *(const v16bf*)(&pm[(wave * 16 + lrow) * 96 + k0 + kh]);
    #pragma unroll
    for (int nt = 0; nt < 4; ++nt) {
      v16bf bf = *(const v16bf*)(&vT[(nt * 16 + lrow) * 96 + k0 + kh]);
      a2[nt] = wmma_bf16(af, bf, a2[nt]);
    }
  }

  // scatter: acc[s,h,n,d] += o, cnt[s,h,n] += keep
  const float keep = s_keep;
  #pragma unroll
  for (int nt = 0; nt < 4; ++nt)
    #pragma unroll
    for (int r = 0; r < 8; ++r) {
      int row = wave * 16 + r + ((lane >> 4) ? 8 : 0);   // query slot 0..63
      int d = nt * 16 + lrow;
      float vv = a2[nt][r] * keep;
      int n = iidx[row];
      atomicAdd(&acc_out[(((size_t)s * HH + h) * NN + n) * DH + d], vv);
    }
  if (tid < KK)
    atomicAdd(&cnt_out[((size_t)s * HH + h) * NN + iidx[tid]], keep);
}

// ---------------------------------------------------------------------------
// finalize: out[s,n,h*DH+d] = acc[s,h,n,d] / max(cnt,1)  -> bf16
// ---------------------------------------------------------------------------
__global__ __launch_bounds__(256)
void finalize_kernel(const float* __restrict__ acc, const float* __restrict__ cnt,
                     __bf16* __restrict__ outn) {
  int i = blockIdx.x * 256 + threadIdx.x;   // over S*H*N*DH = 4M
  const int total = BB * HH * NN * DH;
  if (i >= total) return;
  int d = i % DH;
  int rest = i / DH;
  int n = rest % NN; rest /= NN;
  int h = rest % HH;
  int s = rest / HH;
  float cv = cnt[((size_t)s * HH + h) * NN + n];
  float val = acc[i] / fmaxf(cv, 1.0f);
  outn[((size_t)(s * NN + n)) * DD + h * DH + d] = (__bf16)val;
}

// ---------------------------------------------------------------------------
extern "C" void kernel_launch(void* const* d_in, const int* in_sizes, int n_in,
                              void* d_out, int out_size, void* d_ws, size_t ws_size,
                              hipStream_t stream) {
  const float* x          = (const float*)d_in[0];
  const float* positions  = (const float*)d_in[1];
  const unsigned char* sp = (const unsigned char*)d_in[2];
  const float* w_rms      = (const float*)d_in[3];
  const float* wq         = (const float*)d_in[4];
  const float* wk         = (const float*)d_in[5];
  const float* wv         = (const float*)d_in[6];
  const float* wo         = (const float*)d_in[7];
  const float* w_rope     = (const float*)d_in[8];
  const float* rope_dirs  = (const float*)d_in[9];
  const float* rope_freqs = (const float*)d_in[10];
  const float* in_cent    = (const float*)d_in[11];
  const float* out_cent   = (const float*)d_in[12];
  const float* null_vec   = (const float*)d_in[13];

  char* ws = (char*)d_ws;
  size_t off = 0;
  auto alloc = [&](size_t bytes) -> char* {
    char* p = ws + off;
    off += (bytes + 255) & ~(size_t)255;
    return p;
  };

  const size_t Wn = (size_t)DD * DD;
  __bf16* wq_bf  = (__bf16*)alloc(Wn * 2);
  __bf16* wk_bf  = (__bf16*)alloc(Wn * 2);
  __bf16* wv_bf  = (__bf16*)alloc(Wn * 2);
  __bf16* wo_bf  = (__bf16*)alloc(Wn * 2);
  __bf16* xn_bf  = (__bf16*)alloc((size_t)MM * DD * 2);
  float*  rope   = (float*)alloc((size_t)MM * DD * 4);
  float*  qbuf   = (float*)alloc((size_t)MM * DD * 4);
  float*  kbuf   = (float*)alloc((size_t)MM * DD * 4);
  float*  vbuf   = (float*)alloc((size_t)MM * DD * 4);
  int*    in_idx = (int*)alloc((size_t)BB * HH * CC * KK * 4);
  int*    out_idx= (int*)alloc((size_t)BB * HH * CC * KK * 4);
  float*  accb   = (float*)alloc((size_t)BB * HH * NN * DH * 4);
  float*  cntb   = (float*)alloc((size_t)BB * HH * NN * 4);
  __bf16* outn_bf= (__bf16*)alloc((size_t)MM * DD * 2);

  // 1. weight conversion
  {
    int n = (int)Wn, g = (n + 255) / 256;
    f32_to_bf16_kernel<<<g, 256, 0, stream>>>(wq, wq_bf, n);
    f32_to_bf16_kernel<<<g, 256, 0, stream>>>(wk, wk_bf, n);
    f32_to_bf16_kernel<<<g, 256, 0, stream>>>(wv, wv_bf, n);
    f32_to_bf16_kernel<<<g, 256, 0, stream>>>(wo, wo_bf, n);
  }

  // 2. RMSNorm + RoPE
  prep_kernel<<<MM, 256, 0, stream>>>(x, positions, w_rms, rope_dirs, rope_freqs,
                                      w_rope, xn_bf, rope);

  // 3. q/k/v projections (bf16 WMMA, f32 out, async-LDS staged)
  {
    dim3 grid(MM / 128, DD / 128);
    gemm_bf16_kernel<<<grid, 256, 0, stream>>>(xn_bf, wq_bf, rope, qbuf, MM, DD, DD);
    gemm_bf16_kernel<<<grid, 256, 0, stream>>>(xn_bf, wk_bf, rope, kbuf, MM, DD, DD);
    gemm_bf16_kernel<<<grid, 256, 0, stream>>>(xn_bf, wv_bf, nullptr, vbuf, MM, DD, DD);
  }

  // 4. zero scatter accumulators (ws is poisoned; must re-zero every call)
  {
    int na = BB * HH * NN * DH;
    int nc = BB * HH * NN;
    zero_f32_kernel<<<(na + 255) / 256, 256, 0, stream>>>(accb, na);
    zero_f32_kernel<<<(nc + 255) / 256, 256, 0, stream>>>(cntb, nc);
  }

  // 5. cluster scores + top-K (in from q, out from k)
  {
    int g = BB * HH * CC;
    score_topk_kernel<<<g, 256, 0, stream>>>(qbuf, positions, in_cent, in_idx);
    score_topk_kernel<<<g, 256, 0, stream>>>(kbuf, positions, out_cent, out_idx);
  }

  // 6. routed attention + scatter
  attention_kernel<<<BB * HH * CC, 128, 0, stream>>>(qbuf, kbuf, vbuf, in_idx, out_idx,
                                                     sp, null_vec, accb, cntb);

  // 7. normalize + head-transpose -> bf16
  {
    int total = BB * HH * NN * DH;
    finalize_kernel<<<(total + 255) / 256, 256, 0, stream>>>(accb, cntb, outn_bf);
  }

  // 8. output projection -> d_out (f32)
  {
    dim3 grid(MM / 128, DD / 128);
    gemm_bf16_kernel<<<grid, 256, 0, stream>>>(outn_bf, wo_bf, nullptr, (float*)d_out,
                                               MM, DD, DD);
  }
}
